// SocialEngineeringGenerator_55533927137317
// MI455X (gfx1250) — matrix operand order repaired and checked by reading, not compiled, not run on hardware
//
#include <hip/hip_runtime.h>
#include <hip/hip_bf16.h>
#include <math.h>

// ---------------------------------------------------------------------------
// Types for CDNA5 WMMA (wave32): V_WMMA_F32_16X16X32_BF16
// ---------------------------------------------------------------------------
typedef __attribute__((ext_vector_type(16))) __bf16 v16bf;
typedef __attribute__((ext_vector_type(8)))  float  v8f;

union FragAB {
    v16bf v;
    uint4 u[2];
};

__device__ __forceinline__ unsigned short f2bf(float f) {
    unsigned int u = __float_as_uint(f);
    u += 0x7FFFu + ((u >> 16) & 1u);          // round-to-nearest-even
    return (unsigned short)(u >> 16);
}
__device__ __forceinline__ float bf2f(unsigned short s) {
    return __uint_as_float(((unsigned int)s) << 16);
}
__device__ __forceinline__ float sigm(float x) { return 1.0f / (1.0f + __expf(-x)); }

#define B_        128
#define HID_      512
#define GATES_    2048
#define EMB_      256
#define NOISE_    128
#define VOCAB_    10000
#define K0_       896     // EMB + NOISE + HID (packed [x | h]) for layer 0
#define K1_       1024    // HID + HID for layers >= 1

#define NT_GATES  4       // N tiles per wave in gates GEMM  (2048 = 32 wg * 64)
#define NT_LOGIT  5       // N tiles per wave in logits GEMM (10000 = 125 wg * 80)

// ---------------------------------------------------------------------------
// fp32 -> bf16 convert with strided/offset destination (used to pack [Wih|Whh])
// ---------------------------------------------------------------------------
__global__ void cvt_bf16_k(const float* __restrict__ src, unsigned short* __restrict__ dst,
                           int n, int cols, int dst_ld, int dst_off) {
    int idx = blockIdx.x * blockDim.x + threadIdx.x;
    if (idx >= n) return;
    int r = idx / cols, c = idx - r * cols;
    dst[r * dst_ld + dst_off + c] = f2bf(src[idx]);
}

__global__ void bias_sum_k(const float* __restrict__ a, const float* __restrict__ b,
                           float* __restrict__ out, int n) {
    int i = blockIdx.x * blockDim.x + threadIdx.x;
    if (i < n) out[i] = a[i] + b[i];
}

__global__ void zero_u32_k(unsigned int* __restrict__ p, int n) {
    int i = blockIdx.x * blockDim.x + threadIdx.x;
    if (i < n) p[i] = 0u;
}

// ---------------------------------------------------------------------------
// Build xh0[:, 0:384] = [embedding[tok], noise]  (bf16)
// ---------------------------------------------------------------------------
__global__ void gather_x_k(const unsigned short* __restrict__ emb_bf,
                           const unsigned short* __restrict__ noise_bf,
                           const int* __restrict__ tok, int fixed_tok,
                           unsigned short* __restrict__ xh0) {
    int idx = blockIdx.x * blockDim.x + threadIdx.x;   // 128 * 384
    if (idx >= B_ * (EMB_ + NOISE_)) return;
    int b = idx / (EMB_ + NOISE_);
    int c = idx - b * (EMB_ + NOISE_);
    int t = (fixed_tok >= 0) ? fixed_tok : tok[b];
    unsigned short v = (c < EMB_) ? emb_bf[t * EMB_ + c]
                                  : noise_bf[b * NOISE_ + (c - EMB_)];
    xh0[b * K0_ + c] = v;
}

// ---------------------------------------------------------------------------
// WMMA gates GEMM: gates[128][2048] = xh[128][K] * Wcat[2048][K]^T
// grid.x = 32 (N blocks of 64); block = 256 = 8 waves; wave w -> M rows w*16..
// Each wave: 4 N-tile accumulators, A fragment reused across them.
// ---------------------------------------------------------------------------
__global__ void gemm_gates_k(const unsigned short* __restrict__ xh,
                             const unsigned short* __restrict__ W,
                             float* __restrict__ gates, int K) {
    int lane = threadIdx.x & 31;
    int wave = threadIdx.x >> 5;
    int g    = lane >> 4;        // K-halfgroup select
    int mr   = lane & 15;
    const unsigned short* arow = xh + (wave * 16 + mr) * K;
    const unsigned short* brow[NT_GATES];
#pragma unroll
    for (int nt = 0; nt < NT_GATES; ++nt)
        brow[nt] = W + (blockIdx.x * (16 * NT_GATES) + nt * 16 + mr) * K;

    v8f acc[NT_GATES];
#pragma unroll
    for (int nt = 0; nt < NT_GATES; ++nt) acc[nt] = (v8f){};

    for (int k0 = 0; k0 < K; k0 += 32) {
        FragAB a;
        a.u[0] = *(const uint4*)(arow + k0 + 8 * g);        // K = k0+8g .. +7
        a.u[1] = *(const uint4*)(arow + k0 + 16 + 8 * g);   // K = k0+16+8g .. +7
#pragma unroll
        for (int nt = 0; nt < NT_GATES; ++nt) {
            FragAB b;
            b.u[0] = *(const uint4*)(brow[nt] + k0 + 8 * g);        // K = k0+8g .. +7
            b.u[1] = *(const uint4*)(brow[nt] + k0 + 16 + 8 * g);   // K = k0+16+8g .. +7
            acc[nt] = __builtin_amdgcn_wmma_f32_16x16x32_bf16(
                false, a.v, false, b.v, (short)0, acc[nt], false, false);
        }
    }
    int m0 = wave * 16 + 8 * g;
#pragma unroll
    for (int nt = 0; nt < NT_GATES; ++nt) {
        int ncol = blockIdx.x * (16 * NT_GATES) + nt * 16 + mr;
#pragma unroll
        for (int r = 0; r < 8; ++r)
            gates[(m0 + r) * GATES_ + ncol] = acc[nt][r];
    }
}

// ---------------------------------------------------------------------------
// LSTM elementwise update; writes bf16 h to two destinations
// (own recurrent slot + next layer input slot / hlast)
// ---------------------------------------------------------------------------
__global__ void lstm_elem_k(const float* __restrict__ gates, const float* __restrict__ bsum,
                            float* __restrict__ cstate,
                            unsigned short* __restrict__ d1, int ld1, int off1,
                            unsigned short* __restrict__ d2, int ld2, int off2) {
    int idx = blockIdx.x * blockDim.x + threadIdx.x;   // 128 * 512
    if (idx >= B_ * HID_) return;
    int b = idx >> 9, j = idx & 511;
    const float* gr = gates + b * GATES_;
    float gi = gr[j]            + bsum[j];
    float gf = gr[512 + j]      + bsum[512 + j];
    float gg = gr[1024 + j]     + bsum[1024 + j];
    float go = gr[1536 + j]     + bsum[1536 + j];
    float cn = sigm(gf) * cstate[idx] + sigm(gi) * tanhf(gg);
    float h  = sigm(go) * tanhf(cn);
    cstate[idx] = cn;
    unsigned short hb = f2bf(h);
    d1[b * ld1 + off1 + j] = hb;
    d2[b * ld2 + off2 + j] = hb;
}

// ---------------------------------------------------------------------------
// WMMA vocab projection: out[128][10000] = h[128][512] * Wout[10000][512]^T + bout
// grid.x = 125 (N blocks of 80 = 5 tiles); block = 256 = 8 waves (one M-strip each)
// ---------------------------------------------------------------------------
__global__ void gemm_logits_k(const unsigned short* __restrict__ h,
                              const unsigned short* __restrict__ Wout,
                              const float* __restrict__ bout,
                              float* __restrict__ outbase, long rowstride) {
    int lane = threadIdx.x & 31;
    int wave = threadIdx.x >> 5;
    int g    = lane >> 4;
    int mr   = lane & 15;
    const unsigned short* arow = h + (wave * 16 + mr) * HID_;
    const unsigned short* brow[NT_LOGIT];
#pragma unroll
    for (int nt = 0; nt < NT_LOGIT; ++nt)
        brow[nt] = Wout + (blockIdx.x * (16 * NT_LOGIT) + nt * 16 + mr) * HID_;

    v8f acc[NT_LOGIT];
#pragma unroll
    for (int nt = 0; nt < NT_LOGIT; ++nt) acc[nt] = (v8f){};

#pragma unroll 4
    for (int k0 = 0; k0 < HID_; k0 += 32) {
        FragAB a;
        a.u[0] = *(const uint4*)(arow + k0 + 8 * g);
        a.u[1] = *(const uint4*)(arow + k0 + 16 + 8 * g);
#pragma unroll
        for (int nt = 0; nt < NT_LOGIT; ++nt) {
            FragAB b;
            b.u[0] = *(const uint4*)(brow[nt] + k0 + 8 * g);
            b.u[1] = *(const uint4*)(brow[nt] + k0 + 16 + 8 * g);
            acc[nt] = __builtin_amdgcn_wmma_f32_16x16x32_bf16(
                false, a.v, false, b.v, (short)0, acc[nt], false, false);
        }
    }
    int m0 = wave * 16 + 8 * g;
#pragma unroll
    for (int nt = 0; nt < NT_LOGIT; ++nt) {
        int ncol = blockIdx.x * (16 * NT_LOGIT) + nt * 16 + mr;
        float bb = bout[ncol];
#pragma unroll
        for (int r = 0; r < 8; ++r)
            outbase[(long)(m0 + r) * rowstride + ncol] = acc[nt][r] + bb;
    }
}

// ---------------------------------------------------------------------------
// Row argmax over VOCAB (first-occurrence tie-break, matching jnp.argmax)
// ---------------------------------------------------------------------------
__global__ void argmax_k(const float* __restrict__ base, long rowstride,
                         int* __restrict__ tok) {
    __shared__ float sv[256];
    __shared__ int   si[256];
    const float* row = base + (long)blockIdx.x * rowstride;
    float best = -INFINITY; int bi = 0;
    for (int v = threadIdx.x; v < VOCAB_; v += 256) {
        float x = row[v];
        if (x > best) { best = x; bi = v; }
    }
    sv[threadIdx.x] = best; si[threadIdx.x] = bi;
    __syncthreads();
    for (int s = 128; s > 0; s >>= 1) {
        if (threadIdx.x < s) {
            float xo = sv[threadIdx.x + s]; int io = si[threadIdx.x + s];
            if (xo > sv[threadIdx.x] || (xo == sv[threadIdx.x] && io < si[threadIdx.x])) {
                sv[threadIdx.x] = xo; si[threadIdx.x] = io;
            }
        }
        __syncthreads();
    }
    if (threadIdx.x == 0) tok[blockIdx.x] = si[0];
}

// ---------------------------------------------------------------------------
// Tiny classification heads (4 + 8 + 10 rows, K = 512)
// ---------------------------------------------------------------------------
__global__ void heads_k(const unsigned short* __restrict__ h,
                        const float* __restrict__ uW, const float* __restrict__ ub,
                        const float* __restrict__ eW, const float* __restrict__ eb,
                        const float* __restrict__ pW, const float* __restrict__ pb,
                        float* __restrict__ ou, float* __restrict__ oe, float* __restrict__ op) {
    int b = blockIdx.x, t = threadIdx.x;
    const float *W, *bias; float* o; int row, dim;
    if      (t < 4)  { W = uW; bias = ub; o = ou; row = t;      dim = 4;  }
    else if (t < 12) { W = eW; bias = eb; o = oe; row = t - 4;  dim = 8;  }
    else if (t < 22) { W = pW; bias = pb; o = op; row = t - 12; dim = 10; }
    else return;
    float acc = 0.f;
    for (int k = 0; k < HID_; ++k)
        acc += bf2f(h[b * HID_ + k]) * W[row * HID_ + k];
    o[b * dim + row] = acc + bias[row];
}

// ===========================================================================
// Host side
// ===========================================================================
extern "C" void kernel_launch(void* const* d_in, const int* in_sizes, int n_in,
                              void* d_out, int out_size, void* d_ws, size_t ws_size,
                              hipStream_t stream) {
    (void)in_sizes; (void)n_in; (void)out_size; (void)ws_size;

    const float* noise     = (const float*)d_in[0];
    const float* embedding = (const float*)d_in[1];
    const float* s_Wih0 = (const float*)d_in[2];  const float* s_Whh0 = (const float*)d_in[3];
    const float* s_bih0 = (const float*)d_in[4];  const float* s_bhh0 = (const float*)d_in[5];
    const float* s_Wih1 = (const float*)d_in[6];  const float* s_Whh1 = (const float*)d_in[7];
    const float* s_bih1 = (const float*)d_in[8];  const float* s_bhh1 = (const float*)d_in[9];
    const float* b_Wih0 = (const float*)d_in[10]; const float* b_Whh0 = (const float*)d_in[11];
    const float* b_bih0 = (const float*)d_in[12]; const float* b_bhh0 = (const float*)d_in[13];
    const float* b_Wih1 = (const float*)d_in[14]; const float* b_Whh1 = (const float*)d_in[15];
    const float* b_bih1 = (const float*)d_in[16]; const float* b_bhh1 = (const float*)d_in[17];
    const float* b_Wih2 = (const float*)d_in[18]; const float* b_Whh2 = (const float*)d_in[19];
    const float* b_bih2 = (const float*)d_in[20]; const float* b_bhh2 = (const float*)d_in[21];
    const float* s_Wout = (const float*)d_in[22]; const float* s_bout = (const float*)d_in[23];
    const float* b_Wout = (const float*)d_in[24]; const float* b_bout = (const float*)d_in[25];
    const float* urg_W  = (const float*)d_in[26]; const float* urg_b  = (const float*)d_in[27];
    const float* emo_W  = (const float*)d_in[28]; const float* emo_b  = (const float*)d_in[29];
    const float* pre_W  = (const float*)d_in[30]; const float* pre_b  = (const float*)d_in[31];
    const int SUBJ_LEN = 10, BODY_LEN = 100;   // python-scalar inputs; fixed in reference

    // ---- workspace bump allocator (256B aligned) ----
    char* ws = (char*)d_ws;
    size_t off = 0;
    auto alloc = [&](size_t bytes) -> char* {
        off = (off + 255) & ~(size_t)255;
        char* p = ws + off;
        off += bytes;
        return p;
    };
    unsigned short* emb_bf   = (unsigned short*)alloc((size_t)VOCAB_ * EMB_ * 2);
    unsigned short* noise_bf = (unsigned short*)alloc((size_t)B_ * NOISE_ * 2);
    unsigned short* sW0 = (unsigned short*)alloc((size_t)GATES_ * K0_ * 2);
    unsigned short* sW1 = (unsigned short*)alloc((size_t)GATES_ * K1_ * 2);
    unsigned short* bW0 = (unsigned short*)alloc((size_t)GATES_ * K0_ * 2);
    unsigned short* bW1 = (unsigned short*)alloc((size_t)GATES_ * K1_ * 2);
    unsigned short* bW2 = (unsigned short*)alloc((size_t)GATES_ * K1_ * 2);
    unsigned short* sWoutB = (unsigned short*)alloc((size_t)VOCAB_ * HID_ * 2);
    unsigned short* bWoutB = (unsigned short*)alloc((size_t)VOCAB_ * HID_ * 2);
    float* sbs0 = (float*)alloc(GATES_ * 4);
    float* sbs1 = (float*)alloc(GATES_ * 4);
    float* bbs0 = (float*)alloc(GATES_ * 4);
    float* bbs1 = (float*)alloc(GATES_ * 4);
    float* bbs2 = (float*)alloc(GATES_ * 4);
    float* gates = (float*)alloc((size_t)B_ * GATES_ * 4);
    // state region (zeroed before each generation run)
    size_t state_begin = (off + 255) & ~(size_t)255;
    unsigned short* xh0   = (unsigned short*)alloc((size_t)B_ * K0_ * 2);
    unsigned short* xh1   = (unsigned short*)alloc((size_t)B_ * K1_ * 2);
    unsigned short* xh2   = (unsigned short*)alloc((size_t)B_ * K1_ * 2);
    unsigned short* hlast = (unsigned short*)alloc((size_t)B_ * HID_ * 2);
    float* c0 = (float*)alloc((size_t)B_ * HID_ * 4);
    float* c1 = (float*)alloc((size_t)B_ * HID_ * 4);
    float* c2 = (float*)alloc((size_t)B_ * HID_ * 4);
    int* tok  = (int*)alloc(B_ * 4);
    size_t state_end = (off + 3) & ~(size_t)3;
    int state_words = (int)((state_end - state_begin) / 4);
    unsigned int* state_ptr = (unsigned int*)(ws + state_begin);

    auto cvt = [&](const float* src, unsigned short* dst, int rows, int cols,
                   int ld, int o) {
        int n = rows * cols;
        cvt_bf16_k<<<(n + 255) / 256, 256, 0, stream>>>(src, dst, n, cols, ld, o);
    };
    // ---- weight conversion / packing (bf16) ----
    cvt(embedding, emb_bf, VOCAB_, EMB_, EMB_, 0);
    cvt(noise, noise_bf, B_, NOISE_, NOISE_, 0);
    cvt(s_Wih0, sW0, GATES_, EMB_ + NOISE_, K0_, 0);
    cvt(s_Whh0, sW0, GATES_, HID_, K0_, EMB_ + NOISE_);
    cvt(s_Wih1, sW1, GATES_, HID_, K1_, 0);
    cvt(s_Whh1, sW1, GATES_, HID_, K1_, HID_);
    cvt(b_Wih0, bW0, GATES_, EMB_ + NOISE_, K0_, 0);
    cvt(b_Whh0, bW0, GATES_, HID_, K0_, EMB_ + NOISE_);
    cvt(b_Wih1, bW1, GATES_, HID_, K1_, 0);
    cvt(b_Whh1, bW1, GATES_, HID_, K1_, HID_);
    cvt(b_Wih2, bW2, GATES_, HID_, K1_, 0);
    cvt(b_Whh2, bW2, GATES_, HID_, K1_, HID_);
    cvt(s_Wout, sWoutB, VOCAB_, HID_, HID_, 0);
    cvt(b_Wout, bWoutB, VOCAB_, HID_, HID_, 0);
    bias_sum_k<<<8, 256, 0, stream>>>(s_bih0, s_bhh0, sbs0, GATES_);
    bias_sum_k<<<8, 256, 0, stream>>>(s_bih1, s_bhh1, sbs1, GATES_);
    bias_sum_k<<<8, 256, 0, stream>>>(b_bih0, b_bhh0, bbs0, GATES_);
    bias_sum_k<<<8, 256, 0, stream>>>(b_bih1, b_bhh1, bbs1, GATES_);
    bias_sum_k<<<8, 256, 0, stream>>>(b_bih2, b_bhh2, bbs2, GATES_);

    auto zero_state = [&]() {
        zero_u32_k<<<(state_words + 255) / 256, 256, 0, stream>>>(state_ptr, state_words);
    };
    auto lstm_stack = [&](int L, unsigned short* const* xh, const int* ld,
                          unsigned short* const* Wl, const float* const* bs,
                          float* const* cs) {
        for (int l = 0; l < L; ++l) {
            gemm_gates_k<<<GATES_ / (16 * NT_GATES), 256, 0, stream>>>(
                xh[l], Wl[l], gates, ld[l]);
            unsigned short* d1 = xh[l]; int ld1 = ld[l], off1 = ld[l] - HID_;
            unsigned short* d2; int ld2, off2;
            if (l < L - 1) { d2 = xh[l + 1]; ld2 = ld[l + 1]; off2 = 0; }
            else           { d2 = hlast;     ld2 = HID_;      off2 = 0; }
            lstm_elem_k<<<(B_ * HID_) / 256, 256, 0, stream>>>(
                gates, bs[l], cs[l], d1, ld1, off1, d2, ld2, off2);
        }
    };
    auto generate = [&](int L, unsigned short* const* xh, const int* ld,
                        unsigned short* const* Wl, const float* const* bs,
                        float* const* cs, const unsigned short* WoutB,
                        const float* bout, float* outbase, long rowstride, int steps) {
        zero_state();
        for (int t = 0; t < steps; ++t) {
            gather_x_k<<<(B_ * (EMB_ + NOISE_) + 255) / 256, 256, 0, stream>>>(
                emb_bf, noise_bf, tok, (t == 0) ? 1 : -1, xh[0]);
            lstm_stack(L, xh, ld, Wl, bs, cs);
            float* ob = outbase + (long)t * VOCAB_;
            gemm_logits_k<<<VOCAB_ / (16 * NT_LOGIT), 256, 0, stream>>>(
                hlast, WoutB, bout, ob, rowstride);
            argmax_k<<<B_, 256, 0, stream>>>(ob, rowstride, tok);
        }
    };

    float* out = (float*)d_out;
    const long S_ELEMS = (long)B_ * SUBJ_LEN * VOCAB_;   // 12,800,000
    const long BD_ELEMS = (long)B_ * BODY_LEN * VOCAB_;  // 128,000,000

    // ---- subject (2-layer) ----
    {
        unsigned short* xh[2] = { xh0, xh1 };
        int ld[2] = { K0_, K1_ };
        unsigned short* Wl[2] = { sW0, sW1 };
        const float* bs[2] = { sbs0, sbs1 };
        float* cs[2] = { c0, c1 };
        generate(2, xh, ld, Wl, bs, cs, sWoutB, s_bout,
                 out, (long)SUBJ_LEN * VOCAB_, SUBJ_LEN);
    }
    // ---- body (3-layer) ----
    {
        unsigned short* xh[3] = { xh0, xh1, xh2 };
        int ld[3] = { K0_, K1_, K1_ };
        unsigned short* Wl[3] = { bW0, bW1, bW2 };
        const float* bs[3] = { bbs0, bbs1, bbs2 };
        float* cs[3] = { c0, c1, c2 };
        generate(3, xh, ld, Wl, bs, cs, bWoutB, b_bout,
                 out + S_ELEMS, (long)BODY_LEN * VOCAB_, BODY_LEN);
    }
    // ---- heads: one subject-stack step on token 0 from zero state ----
    {
        zero_state();
        gather_x_k<<<(B_ * (EMB_ + NOISE_) + 255) / 256, 256, 0, stream>>>(
            emb_bf, noise_bf, tok, 0, xh0);
        unsigned short* xh[2] = { xh0, xh1 };
        int ld[2] = { K0_, K1_ };
        unsigned short* Wl[2] = { sW0, sW1 };
        const float* bs[2] = { sbs0, sbs1 };
        float* cs[2] = { c0, c1 };
        lstm_stack(2, xh, ld, Wl, bs, cs);
        float* ou = out + S_ELEMS + BD_ELEMS;
        float* oe = ou + (long)B_ * 4;
        float* op = oe + (long)B_ * 8;
        heads_k<<<B_, 32, 0, stream>>>(hlast, urg_W, urg_b, emo_W, emo_b,
                                       pre_W, pre_b, ou, oe, op);
    }
}
